// MultiHeadAttention_4621384810760
// MI455X (gfx1250) — matrix operand order repaired and checked by reading, not compiled, not run on hardware
//
#include <hip/hip_runtime.h>
#include <hip/hip_bf16.h>
#include <cstdint>

typedef unsigned short ushort_t;
typedef __attribute__((ext_vector_type(16))) __bf16 v16bf;
typedef __attribute__((ext_vector_type(8)))  float  v8f;

#define D_MODEL 1024
#define N_HEADS 16
#define D_K     64
#define SEQ     2048
#define BATCH   4
#define ROWS    (BATCH * SEQ)          // 8192

// ---------- helpers ----------

__device__ __forceinline__ ushort_t f2bf(float f) {
    union { float f; unsigned u; } c; c.f = f;
    unsigned u = c.u;
    unsigned r = u + 0x7FFFu + ((u >> 16) & 1u);   // round-to-nearest-even
    return (ushort_t)(r >> 16);
}

union Frag16 {
    v16bf v;
    uint4 q[2];
};

// 16 contiguous bf16 (B-fragment: lane N = lane%16, K run = 16*hi..+15)
__device__ __forceinline__ v16bf load_b16x16(const ushort_t* p) {
    Frag16 u;
    u.q[0] = *(const uint4*)(p);
    u.q[1] = *(const uint4*)(p + 8);
    return u.v;
}

// A-fragment for 16x32 bf16: elements [0:16] = K 16*hi..+15
__device__ __forceinline__ v16bf load_afrag(const ushort_t* rowp, int hi) {
    Frag16 u;
    u.q[0] = *(const uint4*)(rowp + 16 * hi);
    u.q[1] = *(const uint4*)(rowp + 16 * hi + 8);
    return u.v;
}

__device__ __forceinline__ v8f wmma_bf16(v16bf a, v16bf b, v8f c) {
    return __builtin_amdgcn_wmma_f32_16x16x32_bf16(
        false, a, false, b, (short)0, c, false, false);
}

// async 16B/lane copy global -> LDS (ASYNCcnt-tracked, gfx1250)
__device__ __forceinline__ void async_copy_b128(const ushort_t* gsrc, ushort_t* ldst) {
    unsigned loff = (unsigned)(unsigned long long)(uintptr_t)ldst;  // AS3 offset
    unsigned long long gaddr = (unsigned long long)(uintptr_t)gsrc;
    asm volatile("global_load_async_to_lds_b128 %0, %1, off"
                 :: "v"(loff), "v"(gaddr) : "memory");
}

// ---------- kernel 1: f32 -> bf16 ----------

__global__ void cvt_f32_bf16(const float* __restrict__ src,
                             ushort_t* __restrict__ dst, int n) {
    int i = blockIdx.x * 256 + threadIdx.x;
    if (i < n) dst[i] = f2bf(src[i]);
}

// ---------- kernel 2: fused QKV projection ----------
// out = x @ W^T + b. z = 0:q, 1:k, 2:v(transposed).
// one wave computes a 16(M) x 64(N) tile: A frag reused across 4 WMMAs.
// grid: (ROWS/16/4, D_MODEL/64, 3), block 128 (4 waves)

__global__ __launch_bounds__(128) void qkv_gemm(
    const ushort_t* __restrict__ xb,
    const ushort_t* __restrict__ wqb, const ushort_t* __restrict__ wkb,
    const ushort_t* __restrict__ wvb,
    const float* __restrict__ bq, const float* __restrict__ bk,
    const float* __restrict__ bv,
    ushort_t* __restrict__ qb, ushort_t* __restrict__ kb,
    ushort_t* __restrict__ vtb) {

    const int lane  = threadIdx.x & 31;
    const int widx  = threadIdx.x >> 5;
    const int mtile = blockIdx.x * 4 + widx;
    const int ntile = blockIdx.y;             // 64-wide N block
    const int z     = blockIdx.z;

    const ushort_t* wb   = (z == 0) ? wqb : (z == 1) ? wkb : wvb;
    const float*    bias = (z == 0) ? bq  : (z == 1) ? bk  : bv;

    const int lm = lane & 15;
    const int hi = lane >> 4;

    const ushort_t* arow = xb + (mtile * 16 + lm) * D_MODEL;
    const ushort_t* brow0 = wb + (ntile * 64 +  0 + lm) * D_MODEL + 16 * hi;
    const ushort_t* brow1 = wb + (ntile * 64 + 16 + lm) * D_MODEL + 16 * hi;
    const ushort_t* brow2 = wb + (ntile * 64 + 32 + lm) * D_MODEL + 16 * hi;
    const ushort_t* brow3 = wb + (ntile * 64 + 48 + lm) * D_MODEL + 16 * hi;

    v8f acc0 = {0.f,0.f,0.f,0.f,0.f,0.f,0.f,0.f};
    v8f acc1 = acc0, acc2 = acc0, acc3 = acc0;
#pragma unroll 2
    for (int kk = 0; kk < D_MODEL; kk += 32) {
        v16bf a = load_afrag(arow + kk, hi);
        acc0 = wmma_bf16(a, load_b16x16(brow0 + kk), acc0);
        acc1 = wmma_bf16(a, load_b16x16(brow1 + kk), acc1);
        acc2 = wmma_bf16(a, load_b16x16(brow2 + kk), acc2);
        acc3 = wmma_bf16(a, load_b16x16(brow3 + kk), acc3);
    }

#pragma unroll
    for (int j = 0; j < 4; ++j) {
        const v8f& acc = (j == 0) ? acc0 : (j == 1) ? acc1 : (j == 2) ? acc2 : acc3;
        const int ncol = ntile * 64 + j * 16 + lm;
        const float bcol = bias[ncol];
        const int h = ncol >> 6;          // /64
        const int d = ncol & 63;
#pragma unroll
        for (int e = 0; e < 8; ++e) {
            const int mrow = mtile * 16 + e + 8 * hi;
            const int bidx = mrow >> 11;  // /2048
            const int s    = mrow & 2047;
            const ushort_t bf = f2bf(acc[e] + bcol);
            if (z == 2) {
                vtb[((bidx * N_HEADS + h) * D_K + d) * SEQ + s] = bf;   // [B,H,dk,S]
            } else {
                ushort_t* dst = (z == 0) ? qb : kb;
                dst[((bidx * N_HEADS + h) * SEQ + s) * D_K + d] = bf;   // [B,H,S,dk]
            }
        }
    }
}

// ---------- kernel 3: flash attention ----------
// block = 4 waves sharing one (b,h); each wave owns a 16-row q tile.
// 32-key K/V tiles stream into LDS with async-to-LDS loads, double-buffered.
// grid: (B*H*(SEQ/16)/4), block 128

__global__ __launch_bounds__(128) void attn_kernel(
    const ushort_t* __restrict__ qb, const ushort_t* __restrict__ kb,
    const ushort_t* __restrict__ vtb, ushort_t* __restrict__ ctxb) {

    __shared__ ushort_t ktile[2][32 * 64];   // [key][dk]      4 KB each
    __shared__ ushort_t vtile[2][64 * 32];   // [dk][key]      4 KB each
    __shared__ ushort_t plds[4][16 * 32];    // per-wave P staging

    const int lane   = threadIdx.x & 31;
    const int widx   = threadIdx.x >> 5;
    const int bh     = blockIdx.x >> 5;      // 32 q-groups per (b,h)
    const int qgroup = blockIdx.x & 31;
    const int qtile  = qgroup * 4 + widx;
    const int bidx   = bh >> 4;
    const int h      = bh & 15;

    const ushort_t* qbase = qb  + bh * SEQ * D_K;
    const ushort_t* kbase = kb  + bh * SEQ * D_K;
    const ushort_t* vbase = vtb + bh * D_K * SEQ;

    const int lm = lane & 15;
    const int hi = lane >> 4;

    // q fragments (A-layout), resident for the whole kernel
    const ushort_t* qrow = qbase + (qtile * 16 + lm) * D_K;
    const v16bf qf0 = load_afrag(qrow, hi);
    const v16bf qf1 = load_afrag(qrow + 32, hi);

    v8f acc0 = {0.f,0.f,0.f,0.f,0.f,0.f,0.f,0.f};
    v8f acc1 = acc0, acc2 = acc0, acc3 = acc0;
    float mrow[8], lrow[8];
#pragma unroll
    for (int e = 0; e < 8; ++e) { mrow[e] = -__builtin_inff(); lrow[e] = 0.f; }

    ushort_t* lp = plds[widx];

    // cooperative async tile fetch: 8 KB / 128 threads = 4 x b128 per thread
    auto issue_tile = [&](int buf, int kb0) {
        const int t = threadIdx.x;
#pragma unroll
        for (int j = 0; j < 2; ++j) {
            const int c = j * 128 + t;                       // 16B chunk id
            async_copy_b128(kbase + (kb0 + (c >> 3)) * D_K + ((c & 7) * 8),
                            &ktile[buf][c * 8]);
        }
#pragma unroll
        for (int j = 0; j < 2; ++j) {
            const int c = j * 128 + t;
            async_copy_b128(vbase + (c >> 2) * SEQ + kb0 + ((c & 3) * 8),
                            &vtile[buf][c * 8]);
        }
    };

    issue_tile(0, 0);

    for (int kt = 0; kt < SEQ / 32; ++kt) {
        const int cur = kt & 1;
        const int kb0 = kt * 32;
        if (kt + 1 < SEQ / 32) {
            issue_tile(1 - cur, kb0 + 32);                    // prefetch next
            asm volatile("s_wait_asynccnt 0x4" ::: "memory"); // cur's 4 done
        } else {
            asm volatile("s_wait_asynccnt 0x0" ::: "memory");
        }
        __syncthreads();                                      // tile published

        const ushort_t* ktp = &ktile[cur][0];
        const ushort_t* vtp = &vtile[cur][0];

        // K B-fragments from LDS: N = key, K = dk chunk of 32
        v16bf k00 = load_b16x16(ktp + lm * D_K        + 16 * hi);
        v16bf k01 = load_b16x16(ktp + lm * D_K + 32   + 16 * hi);
        v16bf k10 = load_b16x16(ktp + (16 + lm) * D_K      + 16 * hi);
        v16bf k11 = load_b16x16(ktp + (16 + lm) * D_K + 32 + 16 * hi);

        v8f s0 = {0.f,0.f,0.f,0.f,0.f,0.f,0.f,0.f};
        v8f s1 = s0;
        s0 = wmma_bf16(qf0, k00, s0);
        s0 = wmma_bf16(qf1, k01, s0);
        s1 = wmma_bf16(qf0, k10, s1);
        s1 = wmma_bf16(qf1, k11, s1);

        // online softmax; row e+8*hi lives across the 16-lane group
#pragma unroll
        for (int e = 0; e < 8; ++e) {
            float sa = s0[e] * 0.125f;   // 1/sqrt(64)
            float sb = s1[e] * 0.125f;
            float mx = fmaxf(sa, sb);
#pragma unroll
            for (int off = 1; off < 16; off <<= 1)
                mx = fmaxf(mx, __shfl_xor(mx, off, 32));
            float mn = fmaxf(mrow[e], mx);
            float alpha = __expf(mrow[e] - mn);
            float pa = __expf(sa - mn);
            float pb = __expf(sb - mn);
            float rs = pa + pb;
#pragma unroll
            for (int off = 1; off < 16; off <<= 1)
                rs += __shfl_xor(rs, off, 32);
            lrow[e] = lrow[e] * alpha + rs;
            mrow[e] = mn;
            acc0[e] *= alpha; acc1[e] *= alpha;
            acc2[e] *= alpha; acc3[e] *= alpha;
            const int row = e + 8 * hi;
            lp[row * 32 + lm]      = f2bf(pa);
            lp[row * 32 + lm + 16] = f2bf(pb);
        }

        // same-wave LDS RAW fence for the P re-layout
        asm volatile("s_wait_dscnt 0x0" ::: "memory");

        // P as A-fragment (M=16 q rows, K=32 keys)
        v16bf pf = load_afrag(lp + lm * 32, hi);

        // V B-fragments from LDS: N = dk col, K = key run of 16
#pragma unroll
        for (int nc = 0; nc < 4; ++nc) {
            v16bf vf = load_b16x16(vtp + (nc * 16 + lm) * 32 + 16 * hi);
            if      (nc == 0) acc0 = wmma_bf16(pf, vf, acc0);
            else if (nc == 1) acc1 = wmma_bf16(pf, vf, acc1);
            else if (nc == 2) acc2 = wmma_bf16(pf, vf, acc2);
            else              acc3 = wmma_bf16(pf, vf, acc3);
        }

        __syncthreads();   // protect buf[cur] before kt+1 overwrites it
    }

    // normalize and store ctx as [B,S,D] bf16 (d = h*64 + nc*16 + lm)
#pragma unroll
    for (int e = 0; e < 8; ++e) {
        const float inv = 1.0f / lrow[e];
        const int s = qtile * 16 + e + 8 * hi;
        ushort_t* dst = ctxb + (bidx * SEQ + s) * D_MODEL + h * D_K;
        dst[0 * 16 + lm] = f2bf(acc0[e] * inv);
        dst[1 * 16 + lm] = f2bf(acc1[e] * inv);
        dst[2 * 16 + lm] = f2bf(acc2[e] * inv);
        dst[3 * 16 + lm] = f2bf(acc3[e] * inv);
    }
}

// ---------- kernel 4: output projection (f32 out), 16x64 tile/wave ----------

__global__ __launch_bounds__(128) void outproj_gemm(
    const ushort_t* __restrict__ ctxb, const ushort_t* __restrict__ wob,
    const float* __restrict__ bo, float* __restrict__ out) {

    const int lane  = threadIdx.x & 31;
    const int widx  = threadIdx.x >> 5;
    const int mtile = blockIdx.x * 4 + widx;
    const int ntile = blockIdx.y;
    const int lm = lane & 15;
    const int hi = lane >> 4;

    const ushort_t* arow = ctxb + (mtile * 16 + lm) * D_MODEL;
    const ushort_t* brow0 = wob + (ntile * 64 +  0 + lm) * D_MODEL + 16 * hi;
    const ushort_t* brow1 = wob + (ntile * 64 + 16 + lm) * D_MODEL + 16 * hi;
    const ushort_t* brow2 = wob + (ntile * 64 + 32 + lm) * D_MODEL + 16 * hi;
    const ushort_t* brow3 = wob + (ntile * 64 + 48 + lm) * D_MODEL + 16 * hi;

    v8f acc0 = {0.f,0.f,0.f,0.f,0.f,0.f,0.f,0.f};
    v8f acc1 = acc0, acc2 = acc0, acc3 = acc0;
#pragma unroll 2
    for (int kk = 0; kk < D_MODEL; kk += 32) {
        v16bf a = load_afrag(arow + kk, hi);
        acc0 = wmma_bf16(a, load_b16x16(brow0 + kk), acc0);
        acc1 = wmma_bf16(a, load_b16x16(brow1 + kk), acc1);
        acc2 = wmma_bf16(a, load_b16x16(brow2 + kk), acc2);
        acc3 = wmma_bf16(a, load_b16x16(brow3 + kk), acc3);
    }

#pragma unroll
    for (int j = 0; j < 4; ++j) {
        const v8f& acc = (j == 0) ? acc0 : (j == 1) ? acc1 : (j == 2) ? acc2 : acc3;
        const int ncol = ntile * 64 + j * 16 + lm;
        const float bcol = bo[ncol];
#pragma unroll
        for (int e = 0; e < 8; ++e) {
            const int mrow = mtile * 16 + e + 8 * hi;
            out[mrow * D_MODEL + ncol] = acc[e] + bcol;
        }
    }
}

// ---------- launch ----------

extern "C" void kernel_launch(void* const* d_in, const int* in_sizes, int n_in,
                              void* d_out, int out_size, void* d_ws, size_t ws_size,
                              hipStream_t stream) {
    const float* x  = (const float*)d_in[0];
    const float* wq = (const float*)d_in[1];
    const float* bq = (const float*)d_in[2];
    const float* wk = (const float*)d_in[3];
    const float* bk = (const float*)d_in[4];
    const float* wv = (const float*)d_in[5];
    const float* bv = (const float*)d_in[6];
    const float* wo = (const float*)d_in[7];
    const float* bo = (const float*)d_in[8];

    // workspace layout (bf16 elements)
    ushort_t* ws  = (ushort_t*)d_ws;
    ushort_t* xb   = ws;                                  // 8192*1024
    ushort_t* wqb  = xb   + (size_t)ROWS * D_MODEL;       // 1024*1024
    ushort_t* wkb  = wqb  + (size_t)D_MODEL * D_MODEL;
    ushort_t* wvb  = wkb  + (size_t)D_MODEL * D_MODEL;
    ushort_t* wob  = wvb  + (size_t)D_MODEL * D_MODEL;
    ushort_t* qbuf = wob  + (size_t)D_MODEL * D_MODEL;    // [B,H,S,dk]
    ushort_t* kbuf = qbuf + (size_t)ROWS * D_MODEL;
    ushort_t* vtb  = kbuf + (size_t)ROWS * D_MODEL;       // [B,H,dk,S]
    ushort_t* ctxb = vtb  + (size_t)ROWS * D_MODEL;       // [B,S,D]

    const int nx = ROWS * D_MODEL;       // 8388608
    const int nw = D_MODEL * D_MODEL;    // 1048576

    cvt_f32_bf16<<<nx / 256, 256, 0, stream>>>(x,  xb,  nx);
    cvt_f32_bf16<<<nw / 256, 256, 0, stream>>>(wq, wqb, nw);
    cvt_f32_bf16<<<nw / 256, 256, 0, stream>>>(wk, wkb, nw);
    cvt_f32_bf16<<<nw / 256, 256, 0, stream>>>(wv, wvb, nw);
    cvt_f32_bf16<<<nw / 256, 256, 0, stream>>>(wo, wob, nw);

    // QKV: 512 m-tiles / 4 waves, 16 n-blocks of 64, z = {q,k,v}
    qkv_gemm<<<dim3(ROWS / 16 / 4, D_MODEL / 64, 3), 128, 0, stream>>>(
        xb, wqb, wkb, wvb, bq, bk, bv, qbuf, kbuf, vtb);

    // attention: B*H*(S/16) = 8192 waves, 4 per block sharing K/V tiles
    attn_kernel<<<dim3(BATCH * N_HEADS * (SEQ / 16) / 4), 128, 0, stream>>>(
        qbuf, kbuf, vtb, ctxb);

    outproj_gemm<<<dim3(ROWS / 16 / 4, D_MODEL / 64), 128, 0, stream>>>(
        ctxb, wob, bo, (float*)d_out);
}
